// Attention_3925600109397
// MI455X (gfx1250) — matrix operand order repaired and checked by reading, not compiled, not run on hardware
//
#include <hip/hip_runtime.h>
#include <math.h>

typedef unsigned short u16;
typedef __attribute__((ext_vector_type(16))) __bf16 v16bf;
typedef __attribute__((ext_vector_type(8)))  float  v8f;
typedef __attribute__((ext_vector_type(4)))  unsigned int u32x4;

// float -> bf16 (round-to-nearest-even)
__device__ __forceinline__ u16 f2bf(float f) {
  unsigned int u = __float_as_uint(f);
  u += 0x7FFFu + ((u >> 16) & 1u);
  return (u16)(u >> 16);
}

union FragU { v16bf v; u32x4 u[2]; };

// A-matrix 16x32 bf16 frag: lanes 0-15 hold K=0..15 of their row, lanes 16-31 K=16..31
// caller passes p = base + row*stride + kbase + 16*half
__device__ __forceinline__ v16bf ld_fragA(const u16* p) {
  FragU f;
  f.u[0] = *(const u32x4*)(p);
  f.u[1] = *(const u32x4*)(p + 8);
  return f.v;
}
// B-matrix 32x16 bf16 frag: lanes 0-15 hold K=0..15 of their column, lanes 16-31 K=16..31
// caller passes p = base + col*stride + kbase + 16*half
__device__ __forceinline__ v16bf ld_fragB(const u16* p) {
  FragU f;
  f.u[0] = *(const u32x4*)(p);
  f.u[1] = *(const u32x4*)(p + 8);
  return f.v;
}

__device__ __forceinline__ v8f wmma_bf16(v16bf a, v16bf b, v8f c) {
  return __builtin_amdgcn_wmma_f32_16x16x32_bf16(false, a, false, b, (short)0, c, false, false);
}

__device__ __forceinline__ float half_max(float v) {
  v = fmaxf(v, __shfl_xor(v, 1));
  v = fmaxf(v, __shfl_xor(v, 2));
  v = fmaxf(v, __shfl_xor(v, 4));
  v = fmaxf(v, __shfl_xor(v, 8));
  return v;
}
__device__ __forceinline__ float half_sum(float v) {
  v += __shfl_xor(v, 1);
  v += __shfl_xor(v, 2);
  v += __shfl_xor(v, 4);
  v += __shfl_xor(v, 8);
  return v;
}

__global__ void cvt_bf16(const float* __restrict__ in, u16* __restrict__ out, int n) {
  int i = blockIdx.x * blockDim.x + threadIdx.x;
  if (i < n) out[i] = f2bf(in[i]);
}

// transpose + convert: in [K][N] f32 -> out [N][K] bf16 (32x32 LDS tiles)
__global__ void __launch_bounds__(256) cvt_bf16_t(const float* __restrict__ in,
                                                  u16* __restrict__ out,
                                                  int K, int N) {
  __shared__ u16 t[32][33];
  const int tx = threadIdx.x & 31;
  const int ty = threadIdx.x >> 5;   // 0..7
  const int k0 = blockIdx.y * 32;
  const int n0 = blockIdx.x * 32;
#pragma unroll
  for (int j = 0; j < 4; ++j) {
    int kr = ty * 4 + j;
    t[kr][tx] = f2bf(in[(size_t)(k0 + kr) * N + n0 + tx]);
  }
  __syncthreads();
#pragma unroll
  for (int j = 0; j < 4; ++j) {
    int nr = ty * 4 + j;
    out[(size_t)(n0 + nr) * K + k0 + tx] = t[tx][nr];
  }
}

// ---------------------------------------------------------------------------
// QKV projection: [4096,1024] x Wt[3072][1024] -> scatter Q/K [B*H,N,64], V [B*H,64,N]
// block 256 thr = 8 waves; block tile 128x128; wave tile 64x32; BK=32
// software-pipelined; all LDS staging is vectorized b128 (W pre-transposed)
// ---------------------------------------------------------------------------
__global__ void __launch_bounds__(256) qkv_gemm(const u16* __restrict__ X,
                                                const u16* __restrict__ WT,  // [3072][1024]
                                                u16* __restrict__ Q,
                                                u16* __restrict__ K,
                                                u16* __restrict__ VT) {
  __shared__ u16 sA[128 * 40];
  __shared__ u16 sB[128 * 40];
  const int tid = threadIdx.x;
  const int m0 = blockIdx.y * 128;
  const int n0 = blockIdx.x * 128;
  const int w = tid >> 5, lane = tid & 31, hf = lane >> 4, ln = lane & 15;
  const int wm = (w >> 2) * 64, wn = (w & 3) * 32;

  // staging coordinates (fixed per thread); A and B tiles are both 128 rows x 32 k
  const int sr0 = tid >> 2, sc = tid & 3;
  const int sr1 = (tid + 256) >> 2;

  v8f acc[4][2];
#pragma unroll
  for (int i = 0; i < 4; ++i)
#pragma unroll
    for (int j = 0; j < 2; ++j) acc[i][j] = {};

  // prefetch tile 0
  u32x4 ra0 = *(const u32x4*)(&X[(m0 + sr0) * 1024 + sc * 8]);
  u32x4 ra1 = *(const u32x4*)(&X[(m0 + sr1) * 1024 + sc * 8]);
  u32x4 rb0 = *(const u32x4*)(&WT[(n0 + sr0) * 1024 + sc * 8]);
  u32x4 rb1 = *(const u32x4*)(&WT[(n0 + sr1) * 1024 + sc * 8]);

  for (int kt = 0; kt < 1024; kt += 32) {
    __syncthreads();   // previous compute done reading LDS
    *(u32x4*)(&sA[sr0 * 40 + sc * 8]) = ra0;
    *(u32x4*)(&sA[sr1 * 40 + sc * 8]) = ra1;
    *(u32x4*)(&sB[sr0 * 40 + sc * 8]) = rb0;
    *(u32x4*)(&sB[sr1 * 40 + sc * 8]) = rb1;
    __syncthreads();   // staging visible

    if (kt + 32 < 1024) {  // prefetch next tile (overlaps with WMMA below)
      int kn = kt + 32;
      ra0 = *(const u32x4*)(&X[(m0 + sr0) * 1024 + kn + sc * 8]);
      ra1 = *(const u32x4*)(&X[(m0 + sr1) * 1024 + kn + sc * 8]);
      rb0 = *(const u32x4*)(&WT[(n0 + sr0) * 1024 + kn + sc * 8]);
      rb1 = *(const u32x4*)(&WT[(n0 + sr1) * 1024 + kn + sc * 8]);
    }

    v16bf a[4], b[2];
#pragma unroll
    for (int fm = 0; fm < 4; ++fm) a[fm] = ld_fragA(&sA[(wm + fm * 16 + ln) * 40 + 16 * hf]);
#pragma unroll
    for (int fn = 0; fn < 2; ++fn) b[fn] = ld_fragB(&sB[(wn + fn * 16 + ln) * 40 + 16 * hf]);
#pragma unroll
    for (int fm = 0; fm < 4; ++fm)
#pragma unroll
      for (int fn = 0; fn < 2; ++fn) acc[fm][fn] = wmma_bf16(a[fm], b[fn], acc[fm][fn]);
  }

  // epilogue: `which` is uniform per block (128-col tiles never cross a 1024 boundary)
  const int which = n0 >> 10;                       // 0=Q, 1=K, 2=V
  u16* const dst = (which == 0) ? Q : ((which == 1) ? K : VT);
  const float scl = (which == 0) ? 0.125f : 1.0f;   // 1/sqrt(64) folded into Q
#pragma unroll
  for (int fm = 0; fm < 4; ++fm)
#pragma unroll
    for (int fn = 0; fn < 2; ++fn)
#pragma unroll
      for (int r = 0; r < 8; ++r) {
        int gm = m0 + wm + fm * 16 + r + 8 * hf;
        int gn = n0 + wn + fn * 16 + ln;
        int bb = gm >> 11, nn = gm & 2047;
        int cc = gn & 1023;
        int hh = cc >> 6, dd = cc & 63;
        // Q/K: [b,h][n][d]   V: transposed [b,h][d][n]
        size_t idx = (which == 2)
                         ? ((size_t)((bb * 16 + hh) * 64 + dd)) * 2048 + nn
                         : ((size_t)((bb * 16 + hh) * 2048 + nn)) * 64 + dd;
        dst[idx] = f2bf(acc[fm][fn][r] * scl);
      }
}

// ---------------------------------------------------------------------------
// Flash attention: block = 128 query rows of one (b,h); 8 waves x 16 rows.
// Loop over 64-key tiles; S=Q*K^T (WMMA), online softmax, O += P*V (WMMA).
// Q fragments register-resident; K/V^T tiles prefetched into regs (pipelined).
// ---------------------------------------------------------------------------
__global__ void __launch_bounds__(256) flash_attn(const u16* __restrict__ Q,
                                                  const u16* __restrict__ K,
                                                  const u16* __restrict__ VT,
                                                  u16* __restrict__ Out) {
  __shared__ u16 sQ[128 * 72];
  __shared__ u16 sK[64 * 72];
  __shared__ u16 sVT[64 * 72];     // [d][key]
  __shared__ u16 sP[8 * 16 * 72];  // per-wave 16x64 P tiles
  const int tid = threadIdx.x;
  const int bh = blockIdx.y;   // b*16 + h
  const int qt = blockIdx.x;   // 128-row query tile
  const int w = tid >> 5, lane = tid & 31, hf = lane >> 4, ln = lane & 15;
  const u16* Qp = Q + (size_t)bh * 2048 * 64;
  const u16* Kp = K + (size_t)bh * 2048 * 64;
  const u16* VTp = VT + (size_t)bh * 64 * 2048;   // [64 d][2048 n]

  // stage Q tile [128 x 64]
#pragma unroll
  for (int i = 0; i < 4; ++i) {
    int task = tid + 256 * i;
    int r = task >> 3, c = task & 7;
    *(u32x4*)(&sQ[r * 72 + c * 8]) = *(const u32x4*)(&Qp[(qt * 128 + r) * 64 + c * 8]);
  }
  __syncthreads();
  // Q A-fragments are loop-invariant: keep in registers
  v16bf aq[2];
#pragma unroll
  for (int ks = 0; ks < 2; ++ks)
    aq[ks] = ld_fragA(&sQ[(w * 16 + ln) * 72 + ks * 32 + 16 * hf]);

  v8f acc[4];
#pragma unroll
  for (int i = 0; i < 4; ++i) acc[i] = {};
  float mr[8], lr[8];
#pragma unroll
  for (int r = 0; r < 8; ++r) { mr[r] = -1e30f; lr[r] = 0.f; }
  u16* sPw = sP + w * 16 * 72;

  // staging coordinates: K tile rows = key (64), V^T tile rows = d (64); 8-elem groups
  const int kr0 = tid >> 3, kc0 = tid & 7;
  const int kr1 = (tid + 256) >> 3;

  // prefetch key-tile 0
  u32x4 rk0 = *(const u32x4*)(&Kp[kr0 * 64 + kc0 * 8]);
  u32x4 rk1 = *(const u32x4*)(&Kp[kr1 * 64 + kc0 * 8]);
  u32x4 rv0 = *(const u32x4*)(&VTp[kr0 * 2048 + kc0 * 8]);
  u32x4 rv1 = *(const u32x4*)(&VTp[kr1 * 2048 + kc0 * 8]);

  for (int kt = 0; kt < 2048; kt += 64) {
    __syncthreads();   // previous iteration done reading sK/sVT
    *(u32x4*)(&sK[kr0 * 72 + kc0 * 8]) = rk0;
    *(u32x4*)(&sK[kr1 * 72 + kc0 * 8]) = rk1;
    *(u32x4*)(&sVT[kr0 * 72 + kc0 * 8]) = rv0;
    *(u32x4*)(&sVT[kr1 * 72 + kc0 * 8]) = rv1;
    __syncthreads();

    if (kt + 64 < 2048) {  // prefetch next tile (overlaps with compute below)
      int kn = kt + 64;
      rk0 = *(const u32x4*)(&Kp[(kn + kr0) * 64 + kc0 * 8]);
      rk1 = *(const u32x4*)(&Kp[(kn + kr1) * 64 + kc0 * 8]);
      rv0 = *(const u32x4*)(&VTp[kr0 * 2048 + kn + kc0 * 8]);
      rv1 = *(const u32x4*)(&VTp[kr1 * 2048 + kn + kc0 * 8]);
    }

    // S = Q * K^T : 16 rows x 64 keys per wave
    v8f s[4];
#pragma unroll
    for (int fn = 0; fn < 4; ++fn) s[fn] = {};
#pragma unroll
    for (int ks = 0; ks < 2; ++ks) {
#pragma unroll
      for (int fn = 0; fn < 4; ++fn) {
        v16bf bk = ld_fragB(&sK[(fn * 16 + ln) * 72 + ks * 32 + 16 * hf]);
        s[fn] = wmma_bf16(aq[ks], bk, s[fn]);
      }
    }

    // online softmax per row (row = r + 8*hf; 16 lanes of a half share a row)
#pragma unroll
    for (int r = 0; r < 8; ++r) {
      float mx = fmaxf(fmaxf(s[0][r], s[1][r]), fmaxf(s[2][r], s[3][r]));
      mx = half_max(mx);
      float mnew = fmaxf(mr[r], mx);
      float alpha = __expf(mr[r] - mnew);
      float sum = 0.f;
#pragma unroll
      for (int fn = 0; fn < 4; ++fn) {
        float p = __expf(s[fn][r] - mnew);
        s[fn][r] = p;
        sum += p;
      }
      sum = half_sum(sum);
      lr[r] = lr[r] * alpha + sum;
      mr[r] = mnew;
#pragma unroll
      for (int fd = 0; fd < 4; ++fd) acc[fd][r] *= alpha;
    }

    // re-layout P (D-frag -> A-frag) through wave-private LDS
#pragma unroll
    for (int fn = 0; fn < 4; ++fn)
#pragma unroll
      for (int r = 0; r < 8; ++r)
        sPw[(r + 8 * hf) * 72 + fn * 16 + ln] = f2bf(s[fn][r]);
    asm volatile("s_wait_dscnt 0x0" ::: "memory");

    // O += P * V
#pragma unroll
    for (int ks = 0; ks < 2; ++ks) {
      v16bf ap = ld_fragA(&sPw[ln * 72 + ks * 32 + 16 * hf]);
#pragma unroll
      for (int fd = 0; fd < 4; ++fd) {
        v16bf bv = ld_fragB(&sVT[(fd * 16 + ln) * 72 + ks * 32 + 16 * hf]);
        acc[fd] = wmma_bf16(ap, bv, acc[fd]);
      }
    }
  }

  // normalize + store to [B, N, C] bf16 (head-merged)
  const int b = bh >> 4, h = bh & 15;
#pragma unroll
  for (int r = 0; r < 8; ++r) {
    float inv = 1.f / lr[r];
    int qrow = qt * 128 + w * 16 + r + 8 * hf;
#pragma unroll
    for (int fd = 0; fd < 4; ++fd) {
      int col = h * 64 + fd * 16 + ln;
      Out[((size_t)(b * 2048 + qrow)) * 1024 + col] = f2bf(acc[fd][r] * inv);
    }
  }
}

// ---------------------------------------------------------------------------
// Output projection: [4096,1024] x Wt[1024][1024] + bias -> f32 (pipelined)
// ---------------------------------------------------------------------------
__global__ void __launch_bounds__(256) proj_gemm(const u16* __restrict__ A,
                                                 const u16* __restrict__ WT,  // [1024][1024]
                                                 const float* __restrict__ bias,
                                                 float* __restrict__ Out) {
  __shared__ u16 sA[128 * 40];
  __shared__ u16 sB[128 * 40];
  const int tid = threadIdx.x;
  const int m0 = blockIdx.y * 128;
  const int n0 = blockIdx.x * 128;
  const int w = tid >> 5, lane = tid & 31, hf = lane >> 4, ln = lane & 15;
  const int wm = (w >> 2) * 64, wn = (w & 3) * 32;

  const int sr0 = tid >> 2, sc = tid & 3;
  const int sr1 = (tid + 256) >> 2;

  v8f acc[4][2];
#pragma unroll
  for (int i = 0; i < 4; ++i)
#pragma unroll
    for (int j = 0; j < 2; ++j) acc[i][j] = {};

  u32x4 ra0 = *(const u32x4*)(&A[(m0 + sr0) * 1024 + sc * 8]);
  u32x4 ra1 = *(const u32x4*)(&A[(m0 + sr1) * 1024 + sc * 8]);
  u32x4 rb0 = *(const u32x4*)(&WT[(n0 + sr0) * 1024 + sc * 8]);
  u32x4 rb1 = *(const u32x4*)(&WT[(n0 + sr1) * 1024 + sc * 8]);

  for (int kt = 0; kt < 1024; kt += 32) {
    __syncthreads();
    *(u32x4*)(&sA[sr0 * 40 + sc * 8]) = ra0;
    *(u32x4*)(&sA[sr1 * 40 + sc * 8]) = ra1;
    *(u32x4*)(&sB[sr0 * 40 + sc * 8]) = rb0;
    *(u32x4*)(&sB[sr1 * 40 + sc * 8]) = rb1;
    __syncthreads();

    if (kt + 32 < 1024) {
      int kn = kt + 32;
      ra0 = *(const u32x4*)(&A[(m0 + sr0) * 1024 + kn + sc * 8]);
      ra1 = *(const u32x4*)(&A[(m0 + sr1) * 1024 + kn + sc * 8]);
      rb0 = *(const u32x4*)(&WT[(n0 + sr0) * 1024 + kn + sc * 8]);
      rb1 = *(const u32x4*)(&WT[(n0 + sr1) * 1024 + kn + sc * 8]);
    }

    v16bf a[4], b[2];
#pragma unroll
    for (int fm = 0; fm < 4; ++fm) a[fm] = ld_fragA(&sA[(wm + fm * 16 + ln) * 40 + 16 * hf]);
#pragma unroll
    for (int fn = 0; fn < 2; ++fn) b[fn] = ld_fragB(&sB[(wn + fn * 16 + ln) * 40 + 16 * hf]);
#pragma unroll
    for (int fm = 0; fm < 4; ++fm)
#pragma unroll
      for (int fn = 0; fn < 2; ++fn) acc[fm][fn] = wmma_bf16(a[fm], b[fn], acc[fm][fn]);
  }

#pragma unroll
  for (int fm = 0; fm < 4; ++fm)
#pragma unroll
    for (int fn = 0; fn < 2; ++fn)
#pragma unroll
      for (int r = 0; r < 8; ++r) {
        int gm = m0 + wm + fm * 16 + r + 8 * hf;
        int gn = n0 + wn + fn * 16 + ln;
        Out[(size_t)gm * 1024 + gn] = acc[fm][fn][r] + bias[gn];
      }
}

// ---------------------------------------------------------------------------
extern "C" void kernel_launch(void* const* d_in, const int* in_sizes, int n_in,
                              void* d_out, int out_size, void* d_ws, size_t ws_size,
                              hipStream_t stream) {
  const float* x      = (const float*)d_in[0];   // [2,2048,1024]
  const float* w_qkv  = (const float*)d_in[1];   // [1024,3072]
  const float* w_proj = (const float*)d_in[2];   // [1024,1024]
  const float* b_proj = (const float*)d_in[3];   // [1024]
  float* out = (float*)d_out;                    // [2,2048,1024] f32

  // workspace layout (bf16), total ~48 MB
  u16* xb      = (u16*)d_ws;                     // 4096*1024
  u16* wqkvT   = xb + 4096 * 1024;               // [3072][1024]
  u16* wprojT  = wqkvT + 3072 * 1024;            // [1024][1024]
  u16* qb      = wprojT + 1024 * 1024;           // [32][2048][64]
  u16* kb      = qb + 32 * 2048 * 64;            // [32][2048][64]
  u16* vtb     = kb + 32 * 2048 * 64;            // [32][64][2048]
  u16* ab      = vtb + 32 * 64 * 2048;           // 4096*1024

  cvt_bf16<<<(4096 * 1024) / 256, 256, 0, stream>>>(x, xb, 4096 * 1024);
  cvt_bf16_t<<<dim3(96, 32), 256, 0, stream>>>(w_qkv, wqkvT, 1024, 3072);
  cvt_bf16_t<<<dim3(32, 32), 256, 0, stream>>>(w_proj, wprojT, 1024, 1024);

  qkv_gemm<<<dim3(24, 32), 256, 0, stream>>>(xb, wqkvT, qb, kb, vtb);
  flash_attn<<<dim3(16, 32), 256, 0, stream>>>(qb, kb, vtb, ab);
  proj_gemm<<<dim3(8, 32), 256, 0, stream>>>(ab, wprojT, b_proj, out);
}
